// TransformerBlock_13494787244749
// MI455X (gfx1250) — matrix operand order repaired and checked by reading, not compiled, not run on hardware
//
#include <hip/hip_runtime.h>

#define HWPIX 16384
#define WIMG  128
#define CDIM  192
#define C3    576
#define PTOT  65536

typedef __attribute__((ext_vector_type(16))) __bf16 v16bf;
typedef __attribute__((ext_vector_type(8)))  float  v8f;
typedef __attribute__((ext_vector_type(4)))  unsigned int u32x4;

union V16 { u32x4 q[2]; unsigned short s[16]; v16bf v; };
union U8  { u32x4 q;    unsigned short s[8]; };

__device__ __forceinline__ unsigned short f2bf(float f) {
  unsigned int u = __float_as_uint(f);
  u += 0x7FFFu + ((u >> 16) & 1u);          // round-to-nearest-even
  return (unsigned short)(u >> 16);
}
__device__ __forceinline__ float bf2f(unsigned short h) {
  return __uint_as_float(((unsigned int)h) << 16);
}
__device__ __forceinline__ v8f wmma_bf16(const v16bf& a, const v16bf& b, const v8f& c) {
  return __builtin_amdgcn_wmma_f32_16x16x32_bf16(false, a, false, b, (short)0, c, false, false);
}

// ---------------- one-time weight prep: f32 [Nreal,Kreal] -> bf16 [Npad,Kpad], zero padded ----------------
__global__ void k_wprep(const float* __restrict__ W, unsigned short* __restrict__ Wb,
                        int Kreal, int Nreal, int Kpad) {
  int idx = blockIdx.x * 256 + threadIdx.x;
  int n = idx / Kpad, k = idx - n * Kpad;
  float f = (n < Nreal && k < Kreal) ? W[(size_t)n * Kreal + k] : 0.f;
  Wb[idx] = f2bf(f);
}

// ---------------- NCHW -> [P,C] transpose ----------------
__global__ void k_tin(const float* __restrict__ x, float* __restrict__ xT) {
  __shared__ float tile[32][65];
  int b = blockIdx.z, c0 = blockIdx.y * 32, p0 = blockIdx.x * 64;
  int t = threadIdx.x;
  { int c = t >> 6, p = t & 63;
    for (int cc = c; cc < 32; cc += 4)
      tile[cc][p] = x[((size_t)(b * CDIM + c0 + cc)) * HWPIX + p0 + p]; }
  __syncthreads();
  { int p = t >> 5, c = t & 31;
    for (int pp = p; pp < 64; pp += 8)
      xT[((size_t)(b * HWPIX + p0 + pp)) * CDIM + c0 + c] = tile[c][pp]; }
}

// ---------------- [P,C] -> NCHW (final) ----------------
__global__ void k_tout(const float* __restrict__ y, float* __restrict__ out) {
  __shared__ float tile[64][33];
  int b = blockIdx.z, c0 = blockIdx.y * 32, p0 = blockIdx.x * 64;
  int t = threadIdx.x;
  { int p = t >> 5, c = t & 31;
    for (int pp = p; pp < 64; pp += 8)
      tile[pp][c] = y[((size_t)(b * HWPIX + p0 + pp)) * CDIM + c0 + c]; }
  __syncthreads();
  { int c = t >> 6, p = t & 63;
    for (int cc = c; cc < 32; cc += 4)
      out[((size_t)(b * CDIM + c0 + cc)) * HWPIX + p0 + p] = tile[p][cc]; }
}

// ---------------- channel LayerNorm on [P,192] rows, bf16 out ----------------
__global__ void k_ln(const float* __restrict__ in, const float* __restrict__ w,
                     const float* __restrict__ bias, unsigned short* __restrict__ outb) {
  int wv = threadIdx.x >> 5, lane = threadIdx.x & 31;
  size_t p = (size_t)blockIdx.x * 8 + wv;
  const float* row = in + p * CDIM;
  int base = lane * 6;
  float v[6], s = 0.f, s2 = 0.f;
#pragma unroll
  for (int i = 0; i < 6; i++) { v[i] = row[base + i]; s += v[i]; s2 += v[i] * v[i]; }
#pragma unroll
  for (int m = 16; m >= 1; m >>= 1) { s += __shfl_xor(s, m, 32); s2 += __shfl_xor(s2, m, 32); }
  float mu = s * (1.f / CDIM);
  float var = s2 * (1.f / CDIM) - mu * mu;
  float rs = rsqrtf(var + 1e-5f);
  unsigned short* orow = outb + p * CDIM;
#pragma unroll
  for (int i = 0; i < 6; i++)
    orow[base + i] = f2bf((v[i] - mu) * rs * w[base + i] + bias[base + i]);
}

// ---------------- bf16 WMMA GEMM: out[M,Npad] = A[M,Kpad] * Wb[Npad,Kpad]^T (+res) ----------------
// Block tile: M=256 (8 waves x 2 m-tiles), N=64. Each B fragment from LDS feeds 2 WMMAs.
__global__ void k_gemm(const unsigned short* __restrict__ A, const unsigned short* __restrict__ Wb,
                       float* __restrict__ out, const float* __restrict__ res,
                       int Kpad, int Npad) {
  __shared__ unsigned short wt[64 * 32];
  int t = threadIdx.x, wv = t >> 5, L = t & 31;
  int m0 = blockIdx.x * 256, n0 = blockIdx.y * 64;
  int mrow = m0 + wv * 16 + (L & 15);          // m-tile 0; m-tile 1 is +128
  int kbl = (L >> 4) * 8;
  int nfill = t >> 2, kfill = (t & 3) * 8;
  const unsigned short* wrow = Wb + (size_t)(n0 + nfill) * Kpad + kfill;
  v8f acc0[4] = {}, acc1[4] = {};
  for (int k0 = 0; k0 < Kpad; k0 += 32) {
    __syncthreads();
    *(u32x4*)&wt[nfill * 32 + kfill] = *(const u32x4*)(wrow + k0);
    __syncthreads();
    V16 a0, a1;
    const u32x4* ap0 = (const u32x4*)(A + (size_t)mrow * Kpad + k0 + kbl);
    const u32x4* ap1 = (const u32x4*)(A + (size_t)(mrow + 128) * Kpad + k0 + kbl);
    a0.q[0] = ap0[0]; a0.q[1] = ap0[2];        // K = kbl..+7, kbl+16..+23
    a1.q[0] = ap1[0]; a1.q[1] = ap1[2];
#pragma unroll
    for (int j = 0; j < 4; j++) {
      V16 bmat;
      const u32x4* bp = (const u32x4*)&wt[(j * 16 + (L & 15)) * 32 + (L >> 4) * 8];
      bmat.q[0] = bp[0]; bmat.q[1] = bp[2];
      acc0[j] = wmma_bf16(a0.v, bmat.v, acc0[j]);
      acc1[j] = wmma_bf16(a1.v, bmat.v, acc1[j]);
    }
  }
#pragma unroll
  for (int j = 0; j < 4; j++) {
    int n = n0 + j * 16 + (L & 15);
#pragma unroll
    for (int r = 0; r < 8; r++) {
      int m = m0 + wv * 16 + r + 8 * (L >> 4);
      size_t o0 = (size_t)m * Npad + n;
      size_t o1 = (size_t)(m + 128) * Npad + n;
      float v0 = acc0[j][r], v1 = acc1[j][r];
      if (res) { v0 += res[o0]; v1 += res[o1]; }
      out[o0] = v0;
      out[o1] = v1;
    }
  }
}

// ---------------- depthwise 3x3 on [P,576] + q/k sum-of-squares ----------------
__global__ void k_dwq(const float* __restrict__ in, const float* __restrict__ dw,
                      unsigned short* __restrict__ outb, float* __restrict__ sumsq) {
  __shared__ float ps[384];
  int t = threadIdx.x;
  for (int i = t; i < 384; i += 256) ps[i] = 0.f;
  __syncthreads();
  int P0 = blockIdx.x * 32, b = P0 >> 14;
  for (int idx = t; idx < 32 * C3; idx += 256) {
    int c = idx % C3, pl = idx / C3;
    int pg = P0 + pl, pp = pg & (HWPIX - 1);
    int h = pp >> 7, wq = pp & 127;
    float acc = 0.f;
#pragma unroll
    for (int ky = 0; ky < 3; ky++) {
      int hh = h + ky - 1;
      if ((unsigned)hh < 128u) {
#pragma unroll
        for (int kx = 0; kx < 3; kx++) {
          int ww = wq + kx - 1;
          if ((unsigned)ww < 128u)
            acc += in[((size_t)((b << 14) + (hh << 7) + ww)) * C3 + c] * dw[c * 9 + ky * 3 + kx];
        } } }
    outb[(size_t)pg * C3 + c] = f2bf(acc);
    if (c < 384) atomicAdd(&ps[c], acc * acc);
  }
  __syncthreads();
  for (int i = t; i < 384; i += 256) atomicAdd(&sumsq[b * 384 + i], ps[i]);
}

__global__ void k_scale(const float* __restrict__ sumsq, float* __restrict__ invn) {
  int i = blockIdx.x * 256 + threadIdx.x;
  if (i < 1536) invn[i] = 1.f / fmaxf(sqrtf(sumsq[i]), 1e-12f);
}

// ---------------- channel attention per (b,head): softmax(q̂ k̂ᵀ·T) @ v, all WMMA ----------------
__global__ void k_attn(const unsigned short* __restrict__ qkvd, const float* __restrict__ invn,
                       const float* __restrict__ temp, unsigned short* __restrict__ attnout) {
  __shared__ unsigned short qs[64 * 36];
  __shared__ unsigned short ks[64 * 36];
  __shared__ float smat[64 * 65];
  __shared__ unsigned short pmat[64 * 68];
  int h = blockIdx.x, b = blockIdx.y;
  int t = threadIdx.x, wv = t >> 5, L = t & 31;
  size_t bbase = (size_t)b * HWPIX;
  int qoff = h * 48, koff = 192 + h * 48, voff = 384 + h * 48;
  int mt1 = wv >> 2, nt = wv & 3, mt2 = mt1 + 2;
  int pl = t & 31, cg = t >> 5;
  v8f accA = {}, accB = {};
  for (int p0 = 0; p0 < HWPIX; p0 += 32) {
    __syncthreads();
    { int pg = p0 + pl;
      const unsigned short* row = qkvd + (bbase + pg) * C3;
      V16 uq, uk;
      uq.q[0] = ((const u32x4*)(row + qoff))[cg];
      uk.q[0] = ((const u32x4*)(row + koff))[cg];
#pragma unroll
      for (int i = 0; i < 8; i++) {
        int c = cg * 8 + i;
        int ci = qoff + c; if (ci > 383) ci = 383;
        qs[c * 36 + pl] = f2bf(bf2f(uq.s[i]) * invn[b * 384 + ci]);
        ci = koff + c; if (ci > 383) ci = 383;
        ks[c * 36 + pl] = f2bf(bf2f(uk.s[i]) * invn[b * 384 + ci]);
      } }
    __syncthreads();
    V16 a1, a2, bb;
    { int c = mt1 * 16 + (L & 15), base = c * 36 + (L >> 4) * 8;
#pragma unroll
      for (int i = 0; i < 8; i++) { a1.s[i] = qs[base + i]; a1.s[8 + i] = qs[base + 16 + i]; } }
    { int c = mt2 * 16 + (L & 15), base = c * 36 + (L >> 4) * 8;
#pragma unroll
      for (int i = 0; i < 8; i++) { a2.s[i] = qs[base + i]; a2.s[8 + i] = qs[base + 16 + i]; } }
    { int n = nt * 16 + (L & 15), base = n * 36 + (L >> 4) * 8;
#pragma unroll
      for (int i = 0; i < 16; i++) bb.s[i] = ks[base + i + (i & 8)]; }
    accA = wmma_bf16(a1.v, bb.v, accA);
    accB = wmma_bf16(a2.v, bb.v, accB);
  }
  __syncthreads();
  { float tm = temp[h];
    int n = nt * 16 + (L & 15);
#pragma unroll
    for (int r = 0; r < 8; r++) {
      smat[(mt1 * 16 + r + 8 * (L >> 4)) * 65 + n] = accA[r] * tm;
      smat[(mt2 * 16 + r + 8 * (L >> 4)) * 65 + n] = accB[r] * tm;
    } }
  __syncthreads();
  if (t < 64) {
    int r = t;
    if (r < 48) {
      float mx = -1e30f;
      for (int c = 0; c < 48; c++) mx = fmaxf(mx, smat[r * 65 + c]);
      float sm = 0.f;
      for (int c = 0; c < 48; c++) sm += __expf(smat[r * 65 + c] - mx);
      float inv = 1.f / sm;
      for (int c = 0; c < 48; c++) pmat[r * 68 + c] = f2bf(__expf(smat[r * 65 + c] - mx) * inv);
      for (int c = 48; c < 64; c++) pmat[r * 68 + c] = 0;
    } else {
      for (int c = 0; c < 64; c++) pmat[r * 68 + c] = 0;
    }
  }
  __syncthreads();
  V16 pa[3][2];
#pragma unroll
  for (int mt = 0; mt < 3; mt++)
#pragma unroll
    for (int kk = 0; kk < 2; kk++) {
      int c = mt * 16 + (L & 15), base = c * 68 + kk * 32 + (L >> 4) * 8;
#pragma unroll
      for (int i = 0; i < 8; i++) { pa[mt][kk].s[i] = pmat[base + i]; pa[mt][kk].s[8 + i] = pmat[base + 16 + i]; }
    }
  for (int pt = wv; pt < HWPIX / 16; pt += 8) {
    int p = pt * 16 + (L & 15);
    const unsigned short* vrow = qkvd + (bbase + p) * C3 + voff;
    v8f o[3] = {};
#pragma unroll
    for (int kk = 0; kk < 2; kk++) {
      V16 bv;
      const u32x4* vp = (const u32x4*)vrow;
      int qi = kk * 4 + (L >> 4);
      bv.q[0] = vp[qi]; bv.q[1] = vp[qi + 2];
#pragma unroll
      for (int mt = 0; mt < 3; mt++) o[mt] = wmma_bf16(pa[mt][kk].v, bv.v, o[mt]);
    }
#pragma unroll
    for (int mt = 0; mt < 3; mt++) {
      int coff = h * 48 + mt * 16 + 8 * (L >> 4);
      U8 w8;
#pragma unroll
      for (int r = 0; r < 8; r++) w8.s[r] = f2bf(o[mt][r]);
      *(u32x4*)(attnout + (bbase + p) * CDIM + coff) = w8.q;
    }
  }
}

// ---------------- GDFN depthwise 3x3 on [P,1024] + exact GELU gate -> bf16 [P,512] ----------------
__global__ void k_dwg(const float* __restrict__ in, const float* __restrict__ dw,
                      unsigned short* __restrict__ g) {
  int t = threadIdx.x;
  int P0 = blockIdx.x * 32, b = P0 >> 14;
  for (int idx = t; idx < 32 * 512; idx += 256) {
    int c = idx & 511, pl = idx >> 9;
    int pg = P0 + pl, pp = pg & (HWPIX - 1);
    int h = pp >> 7, wq = pp & 127;
    float gv = 0.f;
    if (c < 510) {
      float d1 = 0.f, d2 = 0.f;
      int c2 = c + 510;
#pragma unroll
      for (int ky = 0; ky < 3; ky++) {
        int hh = h + ky - 1;
        if ((unsigned)hh < 128u) {
#pragma unroll
          for (int kx = 0; kx < 3; kx++) {
            int ww = wq + kx - 1;
            if ((unsigned)ww < 128u) {
              size_t base = ((size_t)((b << 14) + (hh << 7) + ww)) * 1024;
              d1 += in[base + c]  * dw[c * 9 + ky * 3 + kx];
              d2 += in[base + c2] * dw[c2 * 9 + ky * 3 + kx];
            } } } }
      gv = 0.5f * d1 * (1.f + erff(d1 * 0.70710678f)) * d2;
    }
    g[(size_t)pg * 512 + c] = f2bf(gv);
  }
}

extern "C" void kernel_launch(void* const* d_in, const int* in_sizes, int n_in,
                              void* d_out, int out_size, void* d_ws, size_t ws_size,
                              hipStream_t stream) {
  (void)in_sizes; (void)n_in; (void)out_size; (void)ws_size;
  const float* x        = (const float*)d_in[0];
  const float* ln1_w    = (const float*)d_in[1];
  const float* ln1_b    = (const float*)d_in[2];
  const float* qkv_w    = (const float*)d_in[3];
  const float* qkv_dw_w = (const float*)d_in[4];
  const float* temp     = (const float*)d_in[5];
  const float* proj_w   = (const float*)d_in[6];
  const float* ln2_w    = (const float*)d_in[7];
  const float* ln2_b    = (const float*)d_in[8];
  const float* pin_w    = (const float*)d_in[9];
  const float* dw_w     = (const float*)d_in[10];
  const float* pout_w   = (const float*)d_in[11];
  float* out = (float*)d_out;

  char* ws = (char*)d_ws;
  size_t o = 0;
  float*          xT      = (float*)(ws + o);          o += (size_t)PTOT * CDIM * 4;   // 50.3MB
  unsigned short* xn      = (unsigned short*)(ws + o); o += (size_t)PTOT * CDIM * 2;   // 25.2MB
  float*          x2      = (float*)(ws + o);          o += (size_t)PTOT * CDIM * 4;   // 50.3MB
  unsigned short* attnout = (unsigned short*)(ws + o); o += (size_t)PTOT * CDIM * 2;   // 25.2MB
  float*          sumsq   = (float*)(ws + o);          o += 2048 * 4;
  float*          invn    = (float*)(ws + o);          o += 2048 * 4;
  unsigned short* wb_qkv  = (unsigned short*)(ws + o); o += (size_t)576  * 192 * 2;
  unsigned short* wb_proj = (unsigned short*)(ws + o); o += (size_t)192  * 192 * 2;
  unsigned short* wb_pin  = (unsigned short*)(ws + o); o += (size_t)1024 * 192 * 2;
  unsigned short* wb_pout = (unsigned short*)(ws + o); o += (size_t)192  * 512 * 2;
  size_t o_big = (o + 255) & ~(size_t)255;
  float*          qkv     = (float*)(ws + o_big);                                  // 151MB
  unsigned short* qkvd    = (unsigned short*)(ws + o_big + (size_t)PTOT * C3 * 4); // 75.5MB (+slack)
  float*          tbuf    = (float*)(ws + o_big);                                  // aliases qkv/qkvd (dead)
  unsigned short* g       = (unsigned short*)(ws + o_big + (size_t)PTOT * 1024 * 4);
  float*          y3      = xT;                                                    // xT dead after proj

  // one-time bf16 weight prep (tiny; stays resident in L2)
  k_wprep<<<(576 * 192) / 256, 256, 0, stream>>>(qkv_w,  wb_qkv,  192, 576,  192);
  k_wprep<<<(192 * 192) / 256, 256, 0, stream>>>(proj_w, wb_proj, 192, 192,  192);
  k_wprep<<<(1024 * 192) / 256, 256, 0, stream>>>(pin_w, wb_pin,  192, 1020, 192);
  k_wprep<<<(192 * 512) / 256, 256, 0, stream>>>(pout_w, wb_pout, 510, 192,  512);

  k_tin<<<dim3(HWPIX / 64, CDIM / 32, 4), 256, 0, stream>>>(x, xT);
  k_ln<<<PTOT / 8, 256, 0, stream>>>(xT, ln1_w, ln1_b, xn);
  hipMemsetAsync(sumsq, 0, 1536 * 4, stream);
  k_gemm<<<dim3(PTOT / 256, C3 / 64), 256, 0, stream>>>(xn, wb_qkv, qkv, nullptr, 192, 576);
  k_dwq<<<PTOT / 32, 256, 0, stream>>>(qkv, qkv_dw_w, qkvd, sumsq);
  k_scale<<<6, 256, 0, stream>>>(sumsq, invn);
  k_attn<<<dim3(4, 4), 256, 0, stream>>>(qkvd, invn, temp, attnout);
  k_gemm<<<dim3(PTOT / 256, CDIM / 64), 256, 0, stream>>>(attnout, wb_proj, x2, xT, 192, 192);
  k_ln<<<PTOT / 8, 256, 0, stream>>>(x2, ln2_w, ln2_b, xn);
  k_gemm<<<dim3(PTOT / 256, 1024 / 64), 256, 0, stream>>>(xn, wb_pin, tbuf, nullptr, 192, 1024);
  k_dwg<<<PTOT / 32, 256, 0, stream>>>(tbuf, dw_w, g);
  k_gemm<<<dim3(PTOT / 256, CDIM / 64), 256, 0, stream>>>(g, wb_pout, y3, x2, 512, 192);
  k_tout<<<dim3(HWPIX / 64, CDIM / 32, 4), 256, 0, stream>>>(y3, out);
}
